// MMEdgeGCN_88562225643668
// MI455X (gfx1250) — matrix operand order repaired and checked by reading, not compiled, not run on hardware
//
#include <hip/hip_runtime.h>
#include <hip/hip_bf16.h>
#include <stdint.h>

typedef _Float16 v16h __attribute__((ext_vector_type(16)));
typedef float    v8f  __attribute__((ext_vector_type(8)));

#define LN_EPS 1e-5f
#define NPTS 1024
#define NEDGE 4096
#define DDIM 512

// ---------------------------------------------------------------------------
// CDNA5 async global->LDS helpers (ASYNCcnt-tracked; CDNA5 ISA)
// LDS operand = byte offset (low 32 bits of generic pointer to __shared__).
// ---------------------------------------------------------------------------
__device__ __forceinline__ unsigned lds_off(const void* p) {
  return (unsigned)(uintptr_t)p;
}
__device__ __forceinline__ void async_copy_b128(unsigned lds, const float* g) {
  asm volatile("global_load_async_to_lds_b128 %0, %1, off" :: "v"(lds), "v"(g) : "memory");
}
__device__ __forceinline__ void async_copy_b32(unsigned lds, const float* g) {
  asm volatile("global_load_async_to_lds_b32 %0, %1, off" :: "v"(lds), "v"(g) : "memory");
}
__device__ __forceinline__ void wait_async_0()  { asm volatile("s_wait_asynccnt 0x0" ::: "memory"); }
__device__ __forceinline__ void wait_async_12() { asm volatile("s_wait_asynccnt 0xc" ::: "memory"); }

// ---------------------------------------------------------------------------
// LDS-tiled WMMA GEMM: C[M,Nc] = act(A[M,K] @ W[K,Nc] + bias [+ residual])
// Block = 256 threads (8 waves) -> 128x64 C tile; wave -> 32x32 (2x2 wmma tiles).
// Per 32-wide k-step: A(128x32 f32) and B^T(64x32 f32) staged to LDS with
// async-to-LDS (B transposed via per-lane scatter), double buffered.
// A frag (16x32 MxK): lane holds row lane&15; lanes<16: K0..15, lanes>=16: K16..31.
// B frag (32x16 KxN): lane holds col lane&15; lanes<16: K0..15, lanes>=16: K16..31.
// C frag: vgpr r, lane l -> row r + 8*(l>>4), col l&15.
// act: 0 none, 1 relu, 2 sigmoid. M%128==0, Nc%64==0, K%32==0.
// ---------------------------------------------------------------------------
#define BM 128
#define BN 64
#define BK 32

__global__ __launch_bounds__(256)
void gemm_tiled_kernel(const float* __restrict__ A, const float* __restrict__ W,
                       const float* __restrict__ bias, const float* __restrict__ residual,
                       float* __restrict__ C, int M, int K, int Nc, int act)
{
  __shared__ float sA[2][BM * BK];   // [row][k]     16 KB each
  __shared__ float sB[2][BN * BK];   // [n][k] (B^T)  8 KB each
  const int tid  = threadIdx.x;
  const int lane = tid & 31;
  const int wave = tid >> 5;
  const int bn_tiles = Nc / BN;
  const int bm = (blockIdx.x / bn_tiles) * BM;
  const int bn = (blockIdx.x % bn_tiles) * BN;
  const int wm = (wave >> 1) << 5;   // 0,32,64,96
  const int wn = (wave & 1) << 5;    // 0,32
  const int row = lane & 15;
  const int hi  = lane >> 4;

  auto stage = [&](int buf, int kb) {
    // A: 128x32 floats -> 1024 b128 chunks, 4 async instructions per wave
    const unsigned abase = lds_off(&sA[buf][0]);
#pragma unroll
    for (int q = 0; q < 4; ++q) {
      const int chunk = q * 256 + tid;
      const int r = chunk >> 3, c = (chunk & 7) << 2;
      async_copy_b128(abase + (unsigned)((r * BK + c) * 4),
                      A + (size_t)(bm + r) * K + kb + c);
    }
    // B: 32x64 floats, transposed into [n][k]; coalesced global reads
    const unsigned bbase = lds_off(&sB[buf][0]);
#pragma unroll
    for (int q = 0; q < 8; ++q) {
      const int el = q * 256 + tid;
      const int n = el & 63, k = el >> 6;
      async_copy_b32(bbase + (unsigned)((n * BK + k) * 4),
                     W + (size_t)(kb + k) * Nc + bn + n);
    }
  };

  v8f acc[2][2];
#pragma unroll
  for (int ms = 0; ms < 2; ++ms)
#pragma unroll
    for (int ns = 0; ns < 2; ++ns) acc[ms][ns] = (v8f){};

  stage(0, 0);
  const int nsteps = K / BK;
  for (int s = 0; s < nsteps; ++s) {
    const int buf = s & 1;
    if (s + 1 < nsteps) { stage(buf ^ 1, (s + 1) * BK); wait_async_12(); }
    else                { wait_async_0(); }
    __syncthreads();

    v16h af[2];
#pragma unroll
    for (int ms = 0; ms < 2; ++ms) {
      const float* ap = &sA[buf][(wm + ms * 16 + row) * BK + hi * 16];
#pragma unroll
      for (int i = 0; i < 8; ++i) {
        af[ms][i] = (_Float16)ap[i]; af[ms][8 + i] = (_Float16)ap[8 + i];
      }
    }
    v16h bf[2];
#pragma unroll
    for (int ns = 0; ns < 2; ++ns) {
      const float* bp = &sB[buf][(wn + ns * 16 + row) * BK + hi * 16];
#pragma unroll
      for (int i = 0; i < 16; ++i) bf[ns][i] = (_Float16)bp[i];
    }
#pragma unroll
    for (int ms = 0; ms < 2; ++ms)
#pragma unroll
      for (int ns = 0; ns < 2; ++ns)
        acc[ms][ns] = __builtin_amdgcn_wmma_f32_16x16x32_f16(
            false, af[ms], false, bf[ns], (short)0, acc[ms][ns], false, false);
    __syncthreads();
  }

#pragma unroll
  for (int ms = 0; ms < 2; ++ms)
#pragma unroll
    for (int ns = 0; ns < 2; ++ns) {
      const int cn = bn + wn + ns * 16 + row;
#pragma unroll
      for (int r8 = 0; r8 < 8; ++r8) {
        const int cm = bm + wm + ms * 16 + r8 + (hi << 3);
        float v = acc[ms][ns][r8];
        if (bias)     v += bias[cn];
        if (residual) v += residual[(size_t)cm * Nc + cn];
        if (act == 1)      v = fmaxf(v, 0.f);
        else if (act == 2) v = 1.f / (1.f + __expf(-v));
        C[(size_t)cm * Nc + cn] = v;
      }
    }
}

// ---------------------------------------------------------------------------
// Flash-style attention: one wave per (head, 16-q-row tile); streams 32 k-cols
// per iter. S = Q K^T (4 wmma), online softmax (shfl width-16 row reductions),
// P staged via per-wave LDS into A-fragment layout, O += P V (4 wmma).
// Q/K/V/O are [S*, 512] with head h in columns h*64..h*64+63.
// biasb: [H, Sq, Sk] or null; batch: [Sq]==[Sk] ids or null.
// ---------------------------------------------------------------------------
__global__ __launch_bounds__(128)
void attn_kernel(const float* __restrict__ Qm, const float* __restrict__ Km,
                 const float* __restrict__ Vm, const float* __restrict__ biasb,
                 const int* __restrict__ batch, float* __restrict__ Om,
                 int Sq, int Sk, float scale)
{
  __shared__ _Float16 pbuf[4][16 * 32];
  const int lane = threadIdx.x & 31;
  const int wave = threadIdx.x >> 5;
  const int w = blockIdx.x * 4 + wave;
  const int qtiles = Sq >> 4;
  const int h = w / qtiles;
  if (h >= 8) return;
  const int qt = (w % qtiles) << 4;
  const int col = lane & 15;
  const int hi = lane >> 4;
  const int D = DDIM;
  const int hoff = h << 6;
  _Float16* pw = &pbuf[wave][0];

  v16h qa0, qa1;
  {
    const float* qp = Qm + (size_t)(qt + col) * D + hoff + hi * 16;
#pragma unroll
    for (int i = 0; i < 8; ++i) {
      qa0[i] = (_Float16)qp[i];      qa0[8 + i] = (_Float16)qp[8 + i];
      qa1[i] = (_Float16)qp[32 + i]; qa1[8 + i] = (_Float16)qp[40 + i];
    }
  }
  float mrow[8], lrow[8];
#pragma unroll
  for (int r = 0; r < 8; ++r) { mrow[r] = -1e30f; lrow[r] = 0.f; }
  v8f oacc[4];
#pragma unroll
  for (int t = 0; t < 4; ++t) oacc[t] = (v8f){};

  for (int kt = 0; kt < Sk; kt += 32) {
    v8f s0 = {}, s1 = {};
    {
      const float* kp = Km + (size_t)(kt + col) * D + hoff + hi * 16;
      if (kt + 32 < Sk) {
        __builtin_prefetch(kp + (size_t)32 * D, 0, 1);       // global_prefetch_b8
        __builtin_prefetch(Vm + (size_t)(kt + 32 + col) * D + hoff, 0, 1);
      }
      v16h b;
#pragma unroll
      for (int i = 0; i < 16; ++i) b[i] = (_Float16)kp[i];
      s0 = __builtin_amdgcn_wmma_f32_16x16x32_f16(false, qa0, false, b, (short)0, s0, false, false);
#pragma unroll
      for (int i = 0; i < 16; ++i) b[i] = (_Float16)kp[32 + i];
      s0 = __builtin_amdgcn_wmma_f32_16x16x32_f16(false, qa1, false, b, (short)0, s0, false, false);
      const float* kp2 = kp + (size_t)16 * D;
#pragma unroll
      for (int i = 0; i < 16; ++i) b[i] = (_Float16)kp2[i];
      s1 = __builtin_amdgcn_wmma_f32_16x16x32_f16(false, qa0, false, b, (short)0, s1, false, false);
#pragma unroll
      for (int i = 0; i < 16; ++i) b[i] = (_Float16)kp2[32 + i];
      s1 = __builtin_amdgcn_wmma_f32_16x16x32_f16(false, qa1, false, b, (short)0, s1, false, false);
    }
    const int k0 = kt + col, k1 = kt + 16 + col;
#pragma unroll
    for (int r = 0; r < 8; ++r) {
      const int qg = qt + r + (hi << 3);
      float v0 = s0[r] * scale, v1 = s1[r] * scale;
      if (biasb) {
        v0 += biasb[((size_t)h * Sq + qg) * Sk + k0];
        v1 += biasb[((size_t)h * Sq + qg) * Sk + k1];
      }
      if (batch) {
        const int bq = batch[qg];
        if (bq != batch[k0]) v0 = -1e30f;
        if (bq != batch[k1]) v1 = -1e30f;
      }
      float m = fmaxf(v0, v1);
#pragma unroll
      for (int off = 1; off < 16; off <<= 1) m = fmaxf(m, __shfl_xor(m, off, 16));
      const float mnew = fmaxf(mrow[r], m);
      const float corr = __expf(mrow[r] - mnew);
      const float p0 = __expf(v0 - mnew);
      const float p1 = __expf(v1 - mnew);
      float rs = p0 + p1;
#pragma unroll
      for (int off = 1; off < 16; off <<= 1) rs += __shfl_xor(rs, off, 16);
      lrow[r] = lrow[r] * corr + rs;
      mrow[r] = mnew;
#pragma unroll
      for (int t = 0; t < 4; ++t) oacc[t][r] *= corr;
      pw[(r + (hi << 3)) * 32 + col]      = (_Float16)p0;
      pw[(r + (hi << 3)) * 32 + col + 16] = (_Float16)p1;
    }
    // re-shape P (C layout in LDS) into an A fragment; same-wave DS ops are
    // kept in order by HW, so no workgroup barrier needed (per-wave buffer).
    v16h pa;
#pragma unroll
    for (int i = 0; i < 16; ++i)
      pa[i] = pw[col * 32 + (hi << 4) + i];
    const float* vp = Vm + (size_t)(kt + (hi << 4)) * D + hoff;
#pragma unroll
    for (int t = 0; t < 4; ++t) {
      v16h vb;
#pragma unroll
      for (int i = 0; i < 16; ++i) vb[i] = (_Float16)vp[(size_t)i * D + t * 16 + col];
      oacc[t] = __builtin_amdgcn_wmma_f32_16x16x32_f16(false, pa, false, vb, (short)0, oacc[t], false, false);
    }
  }
#pragma unroll
  for (int t = 0; t < 4; ++t)
#pragma unroll
    for (int r = 0; r < 8; ++r) {
      const int qg = qt + r + (hi << 3);
      Om[(size_t)qg * D + hoff + t * 16 + col] = oacc[t][r] / lrow[r];
    }
}

// ---------------------------------------------------------------------------
// Pairwise distance-bias MLP: 1 thread per (i,j) pair; weights staged in LDS.
// ---------------------------------------------------------------------------
__global__ __launch_bounds__(256)
void pair_bias_kernel(const float* __restrict__ ctr, const int* __restrict__ batch,
                      const float* __restrict__ W1, const float* __restrict__ b1,
                      const float* __restrict__ g1, const float* __restrict__ be1,
                      const float* __restrict__ W2, const float* __restrict__ b2,
                      const float* __restrict__ g2, const float* __restrict__ be2,
                      const float* __restrict__ W3, const float* __restrict__ b3,
                      float* __restrict__ bias)
{
  __shared__ float sW1[128], sW2[1024], sW3[256];
  __shared__ float sb1[32], sg1[32], sbe1[32], sb2[32], sg2[32], sbe2[32], sb3[8];
  for (int t = threadIdx.x; t < 128;  t += 256) sW1[t] = W1[t];
  for (int t = threadIdx.x; t < 1024; t += 256) sW2[t] = W2[t];
  for (int t = threadIdx.x; t < 256;  t += 256) sW3[t] = W3[t];
  if (threadIdx.x < 32) {
    sb1[threadIdx.x] = b1[threadIdx.x]; sg1[threadIdx.x] = g1[threadIdx.x];
    sbe1[threadIdx.x] = be1[threadIdx.x]; sb2[threadIdx.x] = b2[threadIdx.x];
    sg2[threadIdx.x] = g2[threadIdx.x]; sbe2[threadIdx.x] = be2[threadIdx.x];
  }
  if (threadIdx.x < 8) sb3[threadIdx.x] = b3[threadIdx.x];
  __syncthreads();

  const int idx = blockIdx.x * 256 + threadIdx.x;
  const int i = idx >> 10, j = idx & 1023;
  const float dx = ctr[j * 3 + 0] - ctr[i * 3 + 0];
  const float dy = ctr[j * 3 + 1] - ctr[i * 3 + 1];
  const float dz = ctr[j * 3 + 2] - ctr[i * 3 + 2];
  const float dd = sqrtf(dx * dx + dy * dy + dz * dz);

  float h1[32];
  float mean = 0.f;
#pragma unroll
  for (int o = 0; o < 32; ++o) {
    float a = sb1[o] + dx * sW1[o] + dy * sW1[32 + o] + dz * sW1[64 + o] + dd * sW1[96 + o];
    h1[o] = a > 0.f ? a : 0.f;
    mean += h1[o];
  }
  mean *= (1.f / 32.f);
  float var = 0.f;
#pragma unroll
  for (int o = 0; o < 32; ++o) { float d = h1[o] - mean; var += d * d; }
  float inv = rsqrtf(var * (1.f / 32.f) + LN_EPS);
#pragma unroll
  for (int o = 0; o < 32; ++o) h1[o] = (h1[o] - mean) * inv * sg1[o] + sbe1[o];

  float h2[32];
  mean = 0.f;
  for (int o = 0; o < 32; ++o) {
    float a = sb2[o];
#pragma unroll
    for (int k = 0; k < 32; ++k) a += h1[k] * sW2[k * 32 + o];
    h2[o] = a > 0.f ? a : 0.f;
    mean += h2[o];
  }
  mean *= (1.f / 32.f);
  var = 0.f;
#pragma unroll
  for (int o = 0; o < 32; ++o) { float d = h2[o] - mean; var += d * d; }
  inv = rsqrtf(var * (1.f / 32.f) + LN_EPS);
#pragma unroll
  for (int o = 0; o < 32; ++o) h2[o] = (h2[o] - mean) * inv * sg2[o] + sbe2[o];

  const bool mv = batch[i] == batch[j];
#pragma unroll
  for (int hh = 0; hh < 8; ++hh) {
    float a = sb3[hh];
#pragma unroll
    for (int k = 0; k < 32; ++k) a += h2[k] * sW3[k * 8 + hh];
    bias[((size_t)hh << 20) + ((size_t)i << 10) + j] = mv ? a : 0.f;
  }
}

// wave-per-row LayerNorm, D = 512
__global__ __launch_bounds__(128)
void ln_kernel(const float* __restrict__ x, const float* __restrict__ g,
               const float* __restrict__ b, float* __restrict__ y, int M)
{
  const int lane = threadIdx.x & 31;
  const int rowi = blockIdx.x * 4 + (threadIdx.x >> 5);
  if (rowi >= M) return;
  const float* xr = x + (size_t)rowi * DDIM;
  float vals[16];
  float s = 0.f;
#pragma unroll
  for (int i = 0; i < 16; ++i) { vals[i] = xr[lane + i * 32]; s += vals[i]; }
#pragma unroll
  for (int off = 1; off < 32; off <<= 1) s += __shfl_xor(s, off, 32);
  const float mean = s * (1.f / DDIM);
  float vs = 0.f;
#pragma unroll
  for (int i = 0; i < 16; ++i) { float d = vals[i] - mean; vs += d * d; }
#pragma unroll
  for (int off = 1; off < 32; off <<= 1) vs += __shfl_xor(vs, off, 32);
  const float inv = rsqrtf(vs * (1.f / DDIM) + LN_EPS);
#pragma unroll
  for (int i = 0; i < 16; ++i) {
    const int c = lane + i * 32;
    y[(size_t)rowi * DDIM + c] = (vals[i] - mean) * inv * g[c] + b[c];
  }
}

// ------------------------- sparse / elementwise ----------------------------
__global__ void deg_kernel(const int* __restrict__ dst, float* __restrict__ deg, int E)
{
  const int e = blockIdx.x * 256 + threadIdx.x;
  if (e < E) atomicAdd(&deg[dst[e]], 1.f);
}
__global__ void dis_kernel(float* __restrict__ deg, int n)
{
  const int i = blockIdx.x * 256 + threadIdx.x;
  if (i < n) deg[i] = rsqrtf(deg[i] + 1.f);   // +1 self loop
}
__global__ void gcn_agg_kernel(const float* __restrict__ xw, const int* __restrict__ src,
                               const int* __restrict__ dst, const float* __restrict__ dis,
                               float* __restrict__ out, int E, int F)
{
  const int t = blockIdx.x * 256 + threadIdx.x;
  const int e = t / F, f = t - e * F;
  if (e < E) {
    const int s = src[e], d = dst[e];
    atomicAdd(&out[(size_t)d * F + f], xw[(size_t)s * F + f] * dis[s] * dis[d]);
  }
}
__global__ void gcn_post_kernel(const float* __restrict__ xw, const float* __restrict__ dis,
                                const float* __restrict__ b, const float* __restrict__ agg,
                                const float* __restrict__ aggregated, float* __restrict__ out,
                                int n, int F)
{
  const int t = blockIdx.x * 256 + threadIdx.x;
  const int i = t / F, f = t - i * F;
  if (i < n) {
    float v = aggregated[t] + xw[t] * dis[i] * dis[i] + b[f];
    v = v > 0.f ? v : 0.f;
    if (agg) v *= agg[t];
    out[t] = v;
  }
}
__global__ void scat_add_kernel(const float* __restrict__ ei, const int* __restrict__ row,
                                const int* __restrict__ col, float* __restrict__ mr,
                                float* __restrict__ mc, int E, int F)
{
  const int t = blockIdx.x * 256 + threadIdx.x;
  const int e = t / F, f = t - e * F;
  if (e < E) {
    const float v = ei[t];
    atomicAdd(&mr[(size_t)row[e] * F + f], v);
    atomicAdd(&mc[(size_t)col[e] * F + f], v);
  }
}
__global__ void count_kernel(const int* __restrict__ row, const int* __restrict__ col,
                             float* __restrict__ cr, float* __restrict__ cc, int E)
{
  const int e = blockIdx.x * 256 + threadIdx.x;
  if (e < E) { atomicAdd(&cr[row[e]], 1.f); atomicAdd(&cc[col[e]], 1.f); }
}
__global__ void aggsig_kernel(const float* __restrict__ mr, const float* __restrict__ mc,
                              const float* __restrict__ cr, const float* __restrict__ cc,
                              float* __restrict__ agg, int n, int F)
{
  const int t = blockIdx.x * 256 + threadIdx.x;
  const int i = t / F;
  if (i < n) {
    const float a = mr[t] / fmaxf(cr[i], 1.f);
    const float b = mc[t] / fmaxf(cc[i], 1.f);
    agg[t] = 1.f / (1.f + __expf(-a * b));
  }
}
__global__ void concat_kernel(const float* __restrict__ ni, const int* __restrict__ row,
                              const int* __restrict__ col, float* __restrict__ ec, int E)
{
  const int t = blockIdx.x * 256 + threadIdx.x;
  const int e = t >> 9, f = t & 511;
  if (e < E)
    ec[t] = (f < 256) ? ni[(size_t)row[e] * 256 + f] : ni[(size_t)col[e] * 256 + f - 256];
}
__global__ void ewmul_kernel(float* __restrict__ x, const float* __restrict__ y, int n)
{
  const int t = blockIdx.x * 256 + threadIdx.x;
  if (t < n) x[t] *= y[t];
}

// ---------------------------------------------------------------------------
// Host side
// ---------------------------------------------------------------------------
static const int kParamSizes[68] = {
  262144,262144,262144,262144,512,512,512,512,512,512,        // cross
  128,1024,256,32,32,8,32,32,32,32,                           // fc
  131072,131072,131072,131072,131072,131072,131072,256,512,256,512,256,256,256, // gcn2d
  131072,131072,131072,131072,131072,131072,131072,256,512,256,512,256,256,256, // gcn3d
  262144,262144,262144,262144,512,512,512,512,512,512,        // rel
  262144,262144,262144,262144,512,512,512,512,512,512         // self
};
enum { PB_CROSS = 0, PB_FC = 10, PB_G2 = 20, PB_G3 = 34, PB_REL = 48, PB_SELF = 58 };
// attention block: +0 Wk, +1 Wo, +2 Wq, +3 Wv, +4 be, +5 bk, +6 bo, +7 bq, +8 bv, +9 g
// gcn block: +0 W1, +1 W2, +2 We1, +3 We2, +4 Wea, +5 Wna, +6 Wnr,
//            +7 b1, +8 b2, +9 be1, +10 be2, +11 bea, +12 bna, +13 bnr

#define OFF_BIG   0u          // 8388608: bias [8,1024,1024]; reused as edge QKVO
#define OFF_QN    8388608u
#define OFF_KN    8912896u
#define OFF_VN    9437184u
#define OFF_AON   9961472u
#define OFF_T1    10485760u
#define OFF_X3    12582912u
#define OFF_X2    13107200u
#define OFF_DIS   13631488u
#define OFF_EI    13632512u
#define OFF_MR    14681088u
#define OFF_MC    14943232u
#define OFF_CR    15205376u
#define OFF_CC    15206400u
#define OFF_AGG   15207424u
#define OFF_XW    15469568u
#define OFF_XAGG  15993856u
#define OFF_XMID  16518144u
#define OFF_NI    16780288u
#define OFF_EC    17042432u
#define OFF_ANI   19139584u
#define OFF_EMID  20188160u
#define OFF_E2T   21236736u

extern "C" void kernel_launch(void* const* d_in, const int* in_sizes, int n_in,
                              void* d_out, int out_size, void* d_ws, size_t ws_size,
                              hipStream_t stream) {
  const float* o3  = (const float*)d_in[0];
  const float* o2  = (const float*)d_in[1];
  const float* e3f = (const float*)d_in[2];
  const float* e2f = (const float*)d_in[3];
  const int*   ei0 = (const int*)d_in[4];
  const int*   ei1 = ((const int*)d_in[4]) + NEDGE;
  const int*   bat = (const int*)d_in[5];
  const float* ctr = (const float*)d_in[6];

  const float* P[68];
  if (n_in >= 7 + 68) {
    for (int i = 0; i < 68; ++i) P[i] = (const float*)d_in[7 + i];
  } else {
    const float* base = (const float*)d_in[7];
    size_t off = 0;
    for (int i = 0; i < 68; ++i) { P[i] = base + off; off += (size_t)kParamSizes[i]; }
  }

  float* ws  = (float*)d_ws;
  float* out = (float*)d_out;
  float* BIAS = ws + OFF_BIG;
  float* oX3  = out;
  float* oE3  = out + 524288;
  float* oX2  = out + 2621440;
  float* oE2  = out + 3145728;

  auto gemm = [&](const float* A, const float* W, const float* b, const float* res,
                  float* C, int M, int K, int Nc, int act) {
    const int blocks = (M / BM) * (Nc / BN);
    gemm_tiled_kernel<<<blocks, 256, 0, stream>>>(A, W, b, res, C, M, K, Nc, act);
  };

  // ---- Stage A: pairwise distance bias [8,1024,1024] ----
  pair_bias_kernel<<<(NPTS * NPTS) / 256, 256, 0, stream>>>(
      ctr, bat, P[PB_FC + 0], P[PB_FC + 3], P[PB_FC + 8], P[PB_FC + 6],
      P[PB_FC + 1], P[PB_FC + 4], P[PB_FC + 9], P[PB_FC + 7],
      P[PB_FC + 2], P[PB_FC + 5], BIAS);

  // ---- Node MHA helper (S = 1024, with bias + mask) ----
  auto mha_node = [&](const float* qin, const float* kvin, int pb, float* xout) {
    gemm(qin,  P[pb + 2], P[pb + 7], nullptr, ws + OFF_QN, NPTS, DDIM, DDIM, 0);
    gemm(kvin, P[pb + 0], P[pb + 5], nullptr, ws + OFF_KN, NPTS, DDIM, DDIM, 0);
    gemm(kvin, P[pb + 3], P[pb + 8], nullptr, ws + OFF_VN, NPTS, DDIM, DDIM, 0);
    attn_kernel<<<(8 * (NPTS / 16)) / 4, 128, 0, stream>>>(
        ws + OFF_QN, ws + OFF_KN, ws + OFF_VN, BIAS, bat, ws + OFF_AON,
        NPTS, NPTS, 0.125f);
    gemm(ws + OFF_AON, P[pb + 1], P[pb + 6], qin, ws + OFF_T1, NPTS, DDIM, DDIM, 0);
    ln_kernel<<<NPTS / 4, 128, 0, stream>>>(ws + OFF_T1, P[pb + 9], P[pb + 4], xout, NPTS);
  };

  mha_node(o3, o3, PB_SELF, ws + OFF_X3);
  mha_node(o2, ws + OFF_X3, PB_CROSS, ws + OFF_X2);

  // ---- shared graph degree normalization ----
  hipMemsetAsync(ws + OFF_DIS, 0, NPTS * sizeof(float), stream);
  deg_kernel<<<NEDGE / 256, 256, 0, stream>>>(ei1, ws + OFF_DIS, NEDGE);
  dis_kernel<<<NPTS / 256, 256, 0, stream>>>(ws + OFF_DIS, NPTS);
  const float* DIS = ws + OFF_DIS;

  // ---- edge GCN helper ----
  auto edge_gcn = [&](const float* xin, const float* ef, int gb, float* xout, float* eout) {
    gemm(ef, P[gb + 4], P[gb + 11], nullptr, ws + OFF_EI, NEDGE, DDIM, 256, 0);
    hipMemsetAsync(ws + OFF_MR, 0, NPTS * 256 * sizeof(float), stream);
    hipMemsetAsync(ws + OFF_MC, 0, NPTS * 256 * sizeof(float), stream);
    hipMemsetAsync(ws + OFF_CR, 0, NPTS * sizeof(float), stream);
    hipMemsetAsync(ws + OFF_CC, 0, NPTS * sizeof(float), stream);
    scat_add_kernel<<<(NEDGE * 256) / 256, 256, 0, stream>>>(
        ws + OFF_EI, ei0, ei1, ws + OFF_MR, ws + OFF_MC, NEDGE, 256);
    count_kernel<<<NEDGE / 256, 256, 0, stream>>>(ei0, ei1, ws + OFF_CR, ws + OFF_CC, NEDGE);
    aggsig_kernel<<<(NPTS * 256) / 256, 256, 0, stream>>>(
        ws + OFF_MR, ws + OFF_MC, ws + OFF_CR, ws + OFF_CC, ws + OFF_AGG, NPTS, 256);
    gemm(xin, P[gb + 0], nullptr, nullptr, ws + OFF_XW, NPTS, DDIM, 256, 0);
    hipMemsetAsync(ws + OFF_XAGG, 0, NPTS * 256 * sizeof(float), stream);
    gcn_agg_kernel<<<(NEDGE * 256) / 256, 256, 0, stream>>>(
        ws + OFF_XW, ei0, ei1, DIS, ws + OFF_XAGG, NEDGE, 256);
    gcn_post_kernel<<<(NPTS * 256) / 256, 256, 0, stream>>>(
        ws + OFF_XW, DIS, P[gb + 7], ws + OFF_AGG, ws + OFF_XAGG, ws + OFF_XMID, NPTS, 256);
    gemm(ws + OFF_XMID, P[gb + 1], nullptr, nullptr, ws + OFF_XW, NPTS, 256, DDIM, 0);
    hipMemsetAsync(ws + OFF_XAGG, 0, NPTS * DDIM * sizeof(float), stream);
    gcn_agg_kernel<<<(NEDGE * DDIM) / 256, 256, 0, stream>>>(
        ws + OFF_XW, ei0, ei1, DIS, ws + OFF_XAGG, NEDGE, DDIM);
    gcn_post_kernel<<<(NPTS * DDIM) / 256, 256, 0, stream>>>(
        ws + OFF_XW, DIS, P[gb + 8], nullptr, ws + OFF_XAGG, xout, NPTS, DDIM);
    gemm(xout, P[gb + 5], P[gb + 12], nullptr, ws + OFF_NI, NPTS, DDIM, 256, 1);
    concat_kernel<<<(NEDGE * DDIM) / 256, 256, 0, stream>>>(
        ws + OFF_NI, ei0, ei1, ws + OFF_EC, NEDGE);
    gemm(ws + OFF_EC, P[gb + 6], P[gb + 13], nullptr, ws + OFF_ANI, NEDGE, DDIM, 256, 2);
    gemm(ef, P[gb + 2], P[gb + 9], nullptr, ws + OFF_EMID, NEDGE, DDIM, 256, 1);
    ewmul_kernel<<<(NEDGE * 256) / 256, 256, 0, stream>>>(
        ws + OFF_EMID, ws + OFF_ANI, NEDGE * 256);
    gemm(ws + OFF_EMID, P[gb + 3], P[gb + 10], nullptr, eout, NEDGE, 256, DDIM, 1);
  };

  edge_gcn(ws + OFF_X3, e3f, PB_G3, oX3, oE3);
  edge_gcn(ws + OFF_X2, e2f, PB_G2, oX2, ws + OFF_E2T);

  // ---- rel MHA on edges (S = 4096, no bias/mask); QKVO reuse BIAS region ----
  {
    float* QE = ws + OFF_BIG;
    float* KE = ws + OFF_BIG + 2097152;
    float* VE = ws + OFF_BIG + 4194304;
    float* AOE = ws + OFF_BIG + 6291456;
    const int pb = PB_REL;
    gemm(ws + OFF_E2T, P[pb + 2], P[pb + 7], nullptr, QE, NEDGE, DDIM, DDIM, 0);
    gemm(oE3, P[pb + 0], P[pb + 5], nullptr, KE, NEDGE, DDIM, DDIM, 0);
    gemm(oE3, P[pb + 3], P[pb + 8], nullptr, VE, NEDGE, DDIM, DDIM, 0);
    attn_kernel<<<(8 * (NEDGE / 16)) / 4, 128, 0, stream>>>(
        QE, KE, VE, nullptr, nullptr, AOE, NEDGE, NEDGE, 0.125f);
    gemm(AOE, P[pb + 1], P[pb + 6], ws + OFF_E2T, ws + OFF_T1, NEDGE, DDIM, DDIM, 0);
    ln_kernel<<<NEDGE / 4, 128, 0, stream>>>(ws + OFF_T1, P[pb + 9], P[pb + 4], oE2, NEDGE);
  }
  (void)in_sizes; (void)n_in; (void)out_size; (void)ws_size;
}
